// SpatialTemporalTransformer_16784732193085
// MI455X (gfx1250) — matrix operand order repaired and checked by reading, not compiled, not run on hardware
//
#include <hip/hip_runtime.h>
#include <math.h>

// ---------------- model dims ----------------
#define Dm   256
#define NHh  8
#define HDh  32
#define FFd  1024
#define Bb   4
#define Ss   16
#define Pp   400
#define Msp  25600      // B*S*P == B*P*S (token rows in both encoders)
#define TOPKk 16

typedef __attribute__((ext_vector_type(16))) _Float16 v16h;
typedef __attribute__((ext_vector_type(8)))  float    v8f;

// =====================================================================
// Weight convert: W[N,K] f32 (torch Linear) -> Wt[K,N] f16 (row-major KxN)
// =====================================================================
__global__ __launch_bounds__(256) void conv_transpose(const float* __restrict__ W,
                                                      _Float16* __restrict__ Wt,
                                                      int N, int K) {
  int idx = blockIdx.x * 256 + threadIdx.x;
  if (idx >= N * K) return;
  int n = idx / K, k = idx - n * K;
  Wt[(size_t)k * N + n] = (_Float16)W[idx];
}

// =====================================================================
// WMMA GEMM: C[M,N] = A[M,K](f32) @ Wt[K,N](f16) + bias, optional relu.
// Block = 4 waves; each wave computes a 16(M) x 64(N) tile (4 accumulators,
// A fragment reused 4x; the 4 waves in a block share identical B rows).
// Requires: M % 64 == 0, N % 64 == 0, K % 32 == 0.
// C row stride = ldc (so q/k/v can pack into one [M,3D] buffer).
// =====================================================================
__global__ __launch_bounds__(128) void gemm_wmma(const float* __restrict__ A,
                                                 const _Float16* __restrict__ Wt,
                                                 const float* __restrict__ bias,
                                                 float* __restrict__ C,
                                                 int M, int N, int K, int ldc, int relu) {
  const int wave = threadIdx.x >> 5;
  const int lane = threadIdx.x & 31;
  const int bm = blockIdx.x * 4 + wave;     // 16-row tile index
  const int bn = blockIdx.y;                // 64-col tile index
  const int m = lane & 15, hi = lane >> 4;

  v8f c0 = {}, c1 = {}, c2 = {}, c3 = {};
  const float4* arow = (const float4*)(A + (size_t)(bm * 16 + m) * K);

  for (int kk = 0; kk < K; kk += 32) {
    // A fragment: 16-bit A 16x32 layout (VGPR0..3: K=koff+0..7, VGPR4..7: K=16+koff..)
    v16h a;
    const int q0 = (kk >> 2) + (hi << 1);
    float4 t0 = arow[q0];
    float4 t1 = arow[q0 + 1];
    float4 u0 = arow[q0 + 4];
    float4 u1 = arow[q0 + 5];
    a[0]  = (_Float16)t0.x; a[1]  = (_Float16)t0.y; a[2]  = (_Float16)t0.z; a[3]  = (_Float16)t0.w;
    a[4]  = (_Float16)t1.x; a[5]  = (_Float16)t1.y; a[6]  = (_Float16)t1.z; a[7]  = (_Float16)t1.w;
    a[8]  = (_Float16)u0.x; a[9]  = (_Float16)u0.y; a[10] = (_Float16)u0.z; a[11] = (_Float16)u0.w;
    a[12] = (_Float16)u1.x; a[13] = (_Float16)u1.y; a[14] = (_Float16)u1.z; a[15] = (_Float16)u1.w;

    // B fragment: lane holds column N = lane%16 (+0/16/32/48), 16 K values in table order
    v16h b0, b1, b2, b3;
    const _Float16* bcol = Wt + (size_t)kk * N + bn * 64 + m;
#pragma unroll
    for (int i = 0; i < 16; ++i) {
      const int kr = (i < 8) ? (hi * 8 + i) : (16 + hi * 8 + i - 8);
      const _Float16* p = bcol + (size_t)kr * N;
      b0[i] = p[0]; b1[i] = p[16]; b2[i] = p[32]; b3[i] = p[48];
    }
    if (kk + 32 < K)
      __builtin_prefetch(Wt + (size_t)(kk + 32 + lane) * N + bn * 64, 0, 1);

    c0 = __builtin_amdgcn_wmma_f32_16x16x32_f16(false, a, false, b0, (short)0, c0, false, false);
    c1 = __builtin_amdgcn_wmma_f32_16x16x32_f16(false, a, false, b1, (short)0, c1, false, false);
    c2 = __builtin_amdgcn_wmma_f32_16x16x32_f16(false, a, false, b2, (short)0, c2, false, false);
    c3 = __builtin_amdgcn_wmma_f32_16x16x32_f16(false, a, false, b3, (short)0, c3, false, false);
  }

  float bv0 = 0.f, bv1 = 0.f, bv2 = 0.f, bv3 = 0.f;
  if (bias) {
    bv0 = bias[bn * 64 + m];
    bv1 = bias[bn * 64 + 16 + m];
    bv2 = bias[bn * 64 + 32 + m];
    bv3 = bias[bn * 64 + 48 + m];
  }
  float* crow = C + (size_t)(bm * 16 + hi * 8) * ldc + bn * 64 + m;
#pragma unroll
  for (int r = 0; r < 8; ++r) {
    float v0 = c0[r] + bv0, v1 = c1[r] + bv1, v2 = c2[r] + bv2, v3 = c3[r] + bv3;
    if (relu) {
      v0 = fmaxf(v0, 0.f); v1 = fmaxf(v1, 0.f);
      v2 = fmaxf(v2, 0.f); v3 = fmaxf(v3, 0.f);
    }
    crow[(size_t)r * ldc +  0] = v0;
    crow[(size_t)r * ldc + 16] = v1;
    crow[(size_t)r * ldc + 32] = v2;
    crow[(size_t)r * ldc + 48] = v3;
  }
}

// =====================================================================
// Attention: qkv[M,3D] rows grouped as numSeq sequences of length L.
// grid = (numSeq, NH, L/16); one wave per block. Dynamic LDS:
//   f32 scores [16][Lpad] then f16 probs [16][Lpad].
// sel != null => sparse union mask (allowed if sel[i] || sel[j]).
// =====================================================================
__global__ __launch_bounds__(32) void attn_kernel(const float* __restrict__ qkv,
                                                  float* __restrict__ out,
                                                  int L, int Lpad,
                                                  const float* __restrict__ sel) {
  extern __shared__ char smem[];
  float*    sc = (float*)smem;
  _Float16* pb = (_Float16*)(smem + (size_t)16 * Lpad * 4);

  const int s = blockIdx.x, h = blockIdx.y, rb = blockIdx.z;
  const int lane = threadIdx.x;
  const int m = lane & 15, hi = lane >> 4;
  const int row0 = s * L + rb * 16;
  const float scale = 0.17677669529663689f;   // 1/sqrt(32)

  // ---- Q fragment (hd == 32 == one K chunk) ----
  v16h aq;
  {
    const float4* qrow = (const float4*)(qkv + (size_t)(row0 + m) * (3 * Dm) + h * HDh);
    float4 t0 = qrow[hi * 2];
    float4 t1 = qrow[hi * 2 + 1];
    float4 u0 = qrow[4 + hi * 2];
    float4 u1 = qrow[5 + hi * 2];
    aq[0]  = (_Float16)t0.x; aq[1]  = (_Float16)t0.y; aq[2]  = (_Float16)t0.z; aq[3]  = (_Float16)t0.w;
    aq[4]  = (_Float16)t1.x; aq[5]  = (_Float16)t1.y; aq[6]  = (_Float16)t1.z; aq[7]  = (_Float16)t1.w;
    aq[8]  = (_Float16)u0.x; aq[9]  = (_Float16)u0.y; aq[10] = (_Float16)u0.z; aq[11] = (_Float16)u0.w;
    aq[12] = (_Float16)u1.x; aq[13] = (_Float16)u1.y; aq[14] = (_Float16)u1.z; aq[15] = (_Float16)u1.w;
  }

  // ---- scores = Q @ K^T / sqrt(hd), tile over columns ----
  for (int jt = 0; jt < L / 16; ++jt) {
    v16h bk;   // B[k][n] = K[jt*16+n][k]; lane supplies n = lane%16, elements over k
    const float* kbase = qkv + (size_t)(s * L + jt * 16 + m) * (3 * Dm) + Dm + h * HDh;
#pragma unroll
    for (int i = 0; i < 16; ++i) bk[i] = (_Float16)kbase[(i < 8) ? (hi * 8 + i) : (16 + hi * 8 + i - 8)];
    v8f c = {};
    c = __builtin_amdgcn_wmma_f32_16x16x32_f16(false, aq, false, bk, (short)0, c, false, false);
    const int col = jt * 16 + m;
    const bool colSel = sel ? (sel[s * L + col] > 0.f) : true;
#pragma unroll
    for (int r = 0; r < 8; ++r) {
      const int rr = r + hi * 8;
      float v = c[r] * scale;
      if (sel) {
        const bool rowSel = sel[s * L + rb * 16 + rr] > 0.f;
        if (!(colSel || rowSel)) v = -1e30f;
      }
      sc[rr * Lpad + col] = v;
    }
  }
  __syncthreads();

  // ---- softmax (row r handled by lanes r and r+16) ----
  {
    const int row = m;
    float mx = -1e30f;
    for (int j = hi; j < L; j += 2) mx = fmaxf(mx, sc[row * Lpad + j]);
    mx = fmaxf(mx, __shfl_xor(mx, 16, 32));
    float sum = 0.f;
    for (int j = hi; j < L; j += 2) {
      float e = expf(sc[row * Lpad + j] - mx);
      sum += e;
      sc[row * Lpad + j] = e;
    }
    sum += __shfl_xor(sum, 16, 32);
    const float inv = 1.f / sum;
    for (int j = hi; j < L; j += 2) pb[row * Lpad + j] = (_Float16)(sc[row * Lpad + j] * inv);
    for (int j = L + hi; j < Lpad; j += 2) pb[row * Lpad + j] = (_Float16)0.f;
  }
  __syncthreads();

  // ---- out = P @ V  (K dim = Lpad, N tiles: d 0..15 and 16..31) ----
  v8f c0 = {}, c1 = {};
  for (int kk = 0; kk < Lpad; kk += 32) {
    v16h ap;
    const _Float16* prow = pb + (size_t)m * Lpad + kk;
#pragma unroll
    for (int j = 0; j < 4; ++j) {
      ap[2 * j]         = prow[hi * 8 + 2 * j];
      ap[2 * j + 1]     = prow[hi * 8 + 2 * j + 1];
      ap[8 + 2 * j]     = prow[16 + hi * 8 + 2 * j];
      ap[8 + 2 * j + 1] = prow[16 + hi * 8 + 2 * j + 1];
    }
    v16h bv0, bv1;
#pragma unroll
    for (int i = 0; i < 16; ++i) {
      const int kIdx = kk + ((i < 8) ? (hi * 8 + i) : (16 + hi * 8 + i - 8));
      if (kIdx < L) {
        const float* vrow = qkv + (size_t)(s * L + kIdx) * (3 * Dm) + 2 * Dm + h * HDh;
        bv0[i] = (_Float16)vrow[m];
        bv1[i] = (_Float16)vrow[16 + m];
      } else {
        bv0[i] = (_Float16)0.f; bv1[i] = (_Float16)0.f;
      }
    }
    c0 = __builtin_amdgcn_wmma_f32_16x16x32_f16(false, ap, false, bv0, (short)0, c0, false, false);
    c1 = __builtin_amdgcn_wmma_f32_16x16x32_f16(false, ap, false, bv1, (short)0, c1, false, false);
  }
#pragma unroll
  for (int r = 0; r < 8; ++r) {
    const size_t orow = (size_t)(row0 + r + hi * 8) * Dm + h * HDh;
    out[orow + m]      = c0[r];
    out[orow + 16 + m] = c1[r];
  }
}

// =====================================================================
// y = LayerNorm(x + h) * g + b   (D = 256, one wave per row)
// =====================================================================
__global__ __launch_bounds__(256) void add_ln(const float* __restrict__ x,
                                              const float* __restrict__ h,
                                              const float* __restrict__ g,
                                              const float* __restrict__ b,
                                              float* __restrict__ y, int M) {
  const int wave = threadIdx.x >> 5, lane = threadIdx.x & 31;
  const int row = blockIdx.x * 8 + wave;
  if (row >= M) return;
  const float* xr = x + (size_t)row * Dm;
  const float* hr = h + (size_t)row * Dm;
  float v[8];
  float s = 0.f;
#pragma unroll
  for (int j = 0; j < 8; ++j) { v[j] = xr[lane * 8 + j] + hr[lane * 8 + j]; s += v[j]; }
  for (int off = 16; off; off >>= 1) s += __shfl_xor(s, off, 32);
  const float mean = s * (1.f / 256.f);
  float var = 0.f;
#pragma unroll
  for (int j = 0; j < 8; ++j) { float t = v[j] - mean; var += t * t; }
  for (int off = 16; off; off >>= 1) var += __shfl_xor(var, off, 32);
  const float inv = rsqrtf(var * (1.f / 256.f) + 1e-5f);
  float* yr = y + (size_t)row * Dm;
#pragma unroll
  for (int j = 0; j < 8; ++j) {
    const int dd = lane * 8 + j;
    yr[dd] = (v[j] - mean) * inv * g[dd] + b[dd];
  }
}

// =====================================================================
// misc elementwise kernels
// =====================================================================
__device__ __forceinline__ float pe_val(int n, int d) {
  const int i = d >> 1;
  const float freq = expf(-(float)(2 * i) * (9.210340371976184f / 256.f));
  const float ang = (float)n * freq;
  return (d & 1) ? cosf(ang) : sinf(ang);
}

// xs[(n*P+p)*D+d] = x[n*P+p]*ew[d] + eb[d] + pe(n,d),  n in [0,B*S)
__global__ __launch_bounds__(256) void embed_pe(const float* __restrict__ x,
                                                const float* __restrict__ ew,
                                                const float* __restrict__ eb,
                                                float* __restrict__ y) {
  int idx = blockIdx.x * 256 + threadIdx.x;
  if (idx >= (Bb * Ss) * Pp * Dm) return;
  const int d = idx & 255;
  const int rest = idx >> 8;
  const int p = rest % Pp;
  const int n = rest / Pp;
  y[idx] = x[n * Pp + p] * ew[d] + eb[d] + pe_val(n, d);
}

// xt[((b*P+p)*S+s)*D+d] = sp[((b*S+s)*P+p)*D+d] + pe(b*P+p, d)
__global__ __launch_bounds__(256) void sp_to_tm(const float* __restrict__ src,
                                                float* __restrict__ dst) {
  int idx = blockIdx.x * 256 + threadIdx.x;
  if (idx >= Msp * Dm) return;
  const int d = idx & 255;
  int rest = idx >> 8;
  const int s = rest & 15; rest >>= 4;
  const int p = rest % Pp;
  const int b = rest / Pp;
  const float v = src[((size_t)(b * Ss + s) * Pp + p) * Dm + d];
  dst[idx] = v + pe_val(b * Pp + p, d);
}

// dst[r*D+d] = src[(r*S + S-1)*D + d],  r in [0, B*P)
__global__ __launch_bounds__(256) void take_last(const float* __restrict__ src,
                                                 float* __restrict__ dst) {
  int idx = blockIdx.x * 256 + threadIdx.x;
  if (idx >= Bb * Pp * Dm) return;
  const int d = idx & 255;
  const int r = idx >> 8;
  dst[idx] = src[((size_t)r * Ss + (Ss - 1)) * Dm + d];
}

// imp[row] = sigmoid(dot(hsel[row,0:128], w) + b)
__global__ __launch_bounds__(256) void sel_imp(const float* __restrict__ hsel,
                                               const float* __restrict__ w,
                                               const float* __restrict__ b,
                                               float* __restrict__ imp, int M) {
  const int wave = threadIdx.x >> 5, lane = threadIdx.x & 31;
  const int row = blockIdx.x * 8 + wave;
  if (row >= M) return;
  float s = 0.f;
  for (int j = lane; j < 128; j += 32) s += hsel[(size_t)row * 128 + j] * w[j];
  for (int off = 16; off; off >>= 1) s += __shfl_xor(s, off, 32);
  imp[row] = 1.f / (1.f + expf(-(s + b[0])));
}

// per-batch top-16 of imp[b*400..] -> sel mask (1.0 at selected positions)
__global__ __launch_bounds__(32) void topk_sel(const float* __restrict__ imp,
                                               float* __restrict__ sel) {
  const int b = blockIdx.x, lane = threadIdx.x;
  const float* ip = imp + b * Pp;
  float* sp = sel + b * Pp;
  for (int j = lane; j < Pp; j += 32) sp[j] = 0.f;
  float vals[13];
  int n = 0;
  for (int j = lane; j < Pp; j += 32) vals[n++] = ip[j];
  for (int t = 0; t < TOPKk; ++t) {
    float best = -1e30f; int bidx = 0x7fffffff;
    for (int q = 0; q < n; ++q) {
      const int j = lane + q * 32;
      if (vals[q] > best || (vals[q] == best && j < bidx)) { best = vals[q]; bidx = j; }
    }
    for (int off = 16; off; off >>= 1) {
      const float ov = __shfl_xor(best, off, 32);
      const int   oi = __shfl_xor(bidx, off, 32);
      if (ov > best || (ov == best && oi < bidx)) { best = ov; bidx = oi; }
    }
    if (lane == (bidx & 31)) {
      sp[bidx] = 1.f;
      const int q = bidx >> 5;
      for (int qq = 0; qq < n; ++qq) if (qq == q) vals[qq] = -1e30f;
    }
  }
}

// pooled[b*D+d] = mean_p so[(b*P+p)*D+d]
__global__ __launch_bounds__(256) void mean_pool(const float* __restrict__ src,
                                                 float* __restrict__ dst) {
  int idx = blockIdx.x * 256 + threadIdx.x;
  if (idx >= Bb * Dm) return;
  const int d = idx & 255, b = idx >> 8;
  float s = 0.f;
  for (int p = 0; p < Pp; ++p) s += src[((size_t)(b * Pp + p)) * Dm + d];
  dst[idx] = s * (1.f / 400.f);
}

__global__ __launch_bounds__(256) void cls_head1(const float* __restrict__ pooled,
                                                 const float* __restrict__ w,
                                                 const float* __restrict__ bias,
                                                 float* __restrict__ hcls) {
  int idx = blockIdx.x * 256 + threadIdx.x;
  if (idx >= Bb * 128) return;
  const int j = idx & 127, b = idx >> 7;
  float s = bias[j];
  for (int k = 0; k < Dm; ++k) s += pooled[b * Dm + k] * w[j * Dm + k];
  hcls[idx] = fmaxf(s, 0.f);
}

__global__ __launch_bounds__(32) void cls_head2(const float* __restrict__ hcls,
                                                const float* __restrict__ w,
                                                const float* __restrict__ bias,
                                                float* __restrict__ out) {
  int idx = threadIdx.x;
  if (idx >= Bb * 3) return;
  const int c = idx % 3, b = idx / 3;
  float s = bias[c];
  for (int k = 0; k < 128; ++k) s += hcls[b * 128 + k] * w[c * 128 + k];
  out[idx] = s;
}

// =====================================================================
// host orchestration
// =====================================================================
extern "C" void kernel_launch(void* const* d_in, const int* in_sizes, int n_in,
                              void* d_out, int out_size, void* d_ws, size_t ws_size,
                              hipStream_t stream) {
  (void)in_sizes; (void)n_in; (void)out_size; (void)ws_size;
  auto IN = [&](int i) { return (const float*)d_in[i]; };

  size_t cur = 0;
  auto alloc = [&](size_t bytes) -> void* {
    void* p = (char*)d_ws + cur;
    cur += (bytes + 255) & ~(size_t)255;
    return p;
  };
  float* actA   = (float*)alloc((size_t)Msp * Dm * 4);
  float* actB   = (float*)alloc((size_t)Msp * Dm * 4);
  float* big    = (float*)alloc((size_t)Msp * FFd * 4);   // qkv / ff / hsel scratch
  float* attnO  = (float*)alloc((size_t)Msp * Dm * 4);
  float* lastB  = (float*)alloc((size_t)Bb * Pp * Dm * 4);
  float* impB   = (float*)alloc((size_t)Bb * Pp * 4);
  float* selB   = (float*)alloc((size_t)Bb * Pp * 4);
  float* pooled = (float*)alloc((size_t)Bb * Dm * 4);
  float* hcls   = (float*)alloc((size_t)Bb * 128 * 4);

  auto conv = [&](const float* W, int N, int K) -> _Float16* {
    _Float16* dst = (_Float16*)alloc((size_t)N * K * 2);
    const int tot = N * K;
    conv_transpose<<<(tot + 255) / 256, 256, 0, stream>>>(W, dst, N, K);
    return dst;
  };

  _Float16 *w_qkv[8], *w_o[8], *w_f1[8], *w_f2[8];
  for (int i = 0; i < 2; ++i) {
    w_qkv[i] = conv(IN(3)  + (size_t)i * 3 * Dm * Dm, 3 * Dm, Dm);
    w_o[i]   = conv(IN(5)  + (size_t)i * Dm * Dm,     Dm,     Dm);
    w_f1[i]  = conv(IN(9)  + (size_t)i * FFd * Dm,    FFd,    Dm);
    w_f2[i]  = conv(IN(11) + (size_t)i * Dm * FFd,    Dm,     FFd);
  }
  for (int i = 0; i < 6; ++i) {
    w_qkv[2 + i] = conv(IN(15) + (size_t)i * 3 * Dm * Dm, 3 * Dm, Dm);
    w_o[2 + i]   = conv(IN(17) + (size_t)i * Dm * Dm,     Dm,     Dm);
    w_f1[2 + i]  = conv(IN(21) + (size_t)i * FFd * Dm,    FFd,    Dm);
    w_f2[2 + i]  = conv(IN(23) + (size_t)i * Dm * FFd,    Dm,     FFd);
  }
  _Float16* w_saq  = conv(IN(27), Dm, Dm);
  _Float16* w_sak  = conv(IN(29), Dm, Dm);
  _Float16* w_sav  = conv(IN(31), Dm, Dm);
  _Float16* w_sao  = conv(IN(33), Dm, Dm);
  _Float16* w_sel1 = conv(IN(35), 128, Dm);

  auto gemm = [&](const float* A, const _Float16* Wt, const float* bias, float* C,
                  int M, int N, int K, int ldc, int relu) {
    gemm_wmma<<<dim3(M / 64, N / 64), 128, 0, stream>>>(A, Wt, bias, C, M, N, K, ldc, relu);
  };

  // embed + spatial PE
  {
    const int tot = (Bb * Ss) * Pp * Dm;
    embed_pe<<<(tot + 255) / 256, 256, 0, stream>>>(IN(0), IN(1), IN(2), actA);
  }

  auto encoder = [&](float* x, float* xalt, int wbase, int nL,
                     const float* bqkv, const float* bo,
                     const float* g1, const float* b1,
                     const float* bf1, const float* bf2,
                     const float* g2, const float* b2,
                     int numSeq, int L) {
    const int Lpad = (L + 31) & ~31;
    const size_t shmem = (size_t)16 * Lpad * 6;   // f32 scores + f16 probs
    for (int i = 0; i < nL; ++i) {
      gemm(x, w_qkv[wbase + i], bqkv + (size_t)i * 3 * Dm, big, Msp, 3 * Dm, Dm, 3 * Dm, 0);
      attn_kernel<<<dim3(numSeq, NHh, L / 16), 32, shmem, stream>>>(big, attnO, L, Lpad,
                                                                    (const float*)nullptr);
      gemm(attnO, w_o[wbase + i], bo + (size_t)i * Dm, xalt, Msp, Dm, Dm, Dm, 0);
      add_ln<<<Msp / 8, 256, 0, stream>>>(x, xalt, g1 + (size_t)i * Dm, b1 + (size_t)i * Dm, x, Msp);
      gemm(x, w_f1[wbase + i], bf1 + (size_t)i * FFd, big, Msp, FFd, Dm, FFd, 1);
      gemm(big, w_f2[wbase + i], bf2 + (size_t)i * Dm, xalt, Msp, Dm, FFd, Dm, 0);
      add_ln<<<Msp / 8, 256, 0, stream>>>(x, xalt, g2 + (size_t)i * Dm, b2 + (size_t)i * Dm, x, Msp);
    }
  };

  // spatial encoder: 64 sequences of length 400
  encoder(actA, actB, 0, 2, IN(4), IN(6), IN(7), IN(8), IN(10), IN(12), IN(13), IN(14),
          Bb * Ss, Pp);

  // transpose to temporal layout + temporal PE
  {
    const int tot = Msp * Dm;
    sp_to_tm<<<(tot + 255) / 256, 256, 0, stream>>>(actA, actB);
  }

  // temporal encoder: 1600 sequences of length 16
  encoder(actB, actA, 2, 6, IN(16), IN(18), IN(19), IN(20), IN(22), IN(24), IN(25), IN(26),
          Bb * Pp, Ss);

  // last timestep -> [B*P, D]
  {
    const int tot = Bb * Pp * Dm;
    take_last<<<(tot + 255) / 256, 256, 0, stream>>>(actB, lastB);
  }

  // selector MLP -> importance -> top-k mask
  gemm(lastB, w_sel1, IN(36), big, Bb * Pp, 128, Dm, 128, 1);
  sel_imp<<<(Bb * Pp) / 8, 256, 0, stream>>>(big, IN(37), IN(38), impB, Bb * Pp);
  topk_sel<<<Bb, 32, 0, stream>>>(impB, selB);

  // sparse attention q/k/v packed into big as [M, 3D]
  gemm(lastB, w_saq, IN(28), big + 0,      Bb * Pp, Dm, Dm, 3 * Dm, 0);
  gemm(lastB, w_sak, IN(30), big + Dm,     Bb * Pp, Dm, Dm, 3 * Dm, 0);
  gemm(lastB, w_sav, IN(32), big + 2 * Dm, Bb * Pp, Dm, Dm, 3 * Dm, 0);
  {
    const int Lpad = (Pp + 31) & ~31;
    attn_kernel<<<dim3(Bb, NHh, Pp / 16), 32, (size_t)16 * Lpad * 6, stream>>>(big, attnO,
                                                                               Pp, Lpad, selB);
  }
  gemm(attnO, w_sao, IN(34), actA, Bb * Pp, Dm, Dm, Dm, 0);

  // pool + classifier head
  mean_pool<<<(Bb * Dm + 255) / 256, 256, 0, stream>>>(actA, pooled);
  cls_head1<<<(Bb * 128 + 255) / 256, 256, 0, stream>>>(pooled, IN(39), IN(40), hcls);
  cls_head2<<<1, 32, 0, stream>>>(hcls, IN(41), IN(42), (float*)d_out);
}
